// Net_30889404793322
// MI455X (gfx1250) — matrix operand order repaired and checked by reading, not compiled, not run on hardware
//
#include <hip/hip_runtime.h>
#include <hip/hip_bf16.h>

typedef __attribute__((ext_vector_type(16))) _Float16 v16h;
typedef __attribute__((ext_vector_type(8)))  float    v8f;

#define BATCH_N   16384
#define KDIM      128
#define HID       256
#define NEDGES    253

// Branch-free activations on hardware transcendentals (v_exp_f32 / v_rcp_f32).
__device__ __forceinline__ float sigmoidf_(float x) {
    return __builtin_amdgcn_rcpf(1.0f + __expf(-x));
}
// tanh(x) = 1 - 2/(1 + exp(2x)); saturates exactly to +/-1 when v_exp_f32
// over/underflows, no branches, EXEC stays uniform.
__device__ __forceinline__ float tanhf_(float x) {
    return 1.0f - 2.0f * __builtin_amdgcn_rcpf(1.0f + __expf(2.0f * x));
}

// ---------------------------------------------------------------------------
// Kernel 0: convert Wu (HID x KDIM) and Wi (HID x KDIM) to f16, same layout.
// B-matrix reads in the GEMM are two 8-f16 runs (k +0..7, +16..23) from a row of these.
// ---------------------------------------------------------------------------
__global__ __launch_bounds__(256) void cvt_kernel(const float* __restrict__ Wu,
                                                  const float* __restrict__ Wi,
                                                  _Float16* __restrict__ Wuh,
                                                  _Float16* __restrict__ Wih) {
    int i = blockIdx.x * 256 + threadIdx.x;
    if (i < HID * KDIM) {
        Wuh[i] = (_Float16)Wu[i];
        Wih[i] = (_Float16)Wi[i];
    }
}

// ---------------------------------------------------------------------------
// Kernel 1: per-batch-row posterior scan (126 sequential steps) fused with
// the item_know elementwise products. Emits f16 A-matrices for both GEMMs
// plus the per-row sigmoided discrimination scalar.
// Note: the reference only uses priori[:, 0] — no need to load full rows.
// ---------------------------------------------------------------------------
__global__ __launch_bounds__(256) void prep_kernel(
    const int*   __restrict__ user_ids,
    const int*   __restrict__ item_ids,
    const float* __restrict__ item_know,
    const float* __restrict__ priori,
    const float* __restrict__ condi_p,
    const float* __restrict__ condi_n,
    const float* __restrict__ item_diff_w,
    const float* __restrict__ item_disc_w,
    _Float16* __restrict__ Au,
    _Float16* __restrict__ Ai,
    float* __restrict__ idisc) {
    int i = blockIdx.x * 256 + threadIdx.x;
    if (i >= BATCH_N) return;

    const int u  = user_ids[i];
    const int it = item_ids[i];

    const float* cp  = condi_p     + (size_t)u  * NEDGES;
    const float* cn  = condi_n     + (size_t)u  * NEDGES;
    const float* ik  = item_know   + (size_t)i  * KDIM;
    const float* idw = item_diff_w + (size_t)it * KDIM;
    _Float16* aur = Au + (size_t)i * KDIM;
    _Float16* air = Ai + (size_t)i * KDIM;

    // posterior scan: p_k from p_{k-1}, p_{k-2}
    float p2 = sigmoidf_(priori[(size_t)u * KDIM]);                     // p0
    float p1 = sigmoidf_(cp[0]) * p2 + sigmoidf_(cn[0]) * (1.0f - p2);  // p1
    aur[0] = (_Float16)(p2 * ik[0]);
    aur[1] = (_Float16)(p1 * ik[1]);
    for (int t = 0; t < KDIM - 2; ++t) {
        float ta = __builtin_amdgcn_sqrtf(sigmoidf_(cp[KDIM - 1 + t])) * p2 +
                   __builtin_amdgcn_sqrtf(sigmoidf_(cn[KDIM - 1 + t])) * (1.0f - p2);
        float tb = __builtin_amdgcn_sqrtf(sigmoidf_(cp[1 + t])) * p1 +
                   __builtin_amdgcn_sqrtf(sigmoidf_(cn[1 + t])) * (1.0f - p1);
        float pk = ta * tb;
        aur[t + 2] = (_Float16)(pk * ik[t + 2]);
        p2 = p1;
        p1 = pk;
    }

    for (int k = 0; k < KDIM; ++k)
        air[k] = (_Float16)(sigmoidf_(idw[k]) * ik[k]);

    idisc[i] = sigmoidf_(item_disc_w[it]);
}

// ---------------------------------------------------------------------------
// Kernel 2: fused dual-GEMM + nonlinearities + row dot + final sigmoid.
// One wave handles a 16-row batch tile: 16 N-tiles x 4 K-steps x 2 matrices
// = 128 v_wmma_f32_16x16x32_f16 per wave.
//
// Fragment layouts match the ISA VGPR striping:
//   A (16x32 f16): lane = half*16 + m ; vector elem e<8  -> K = half*8 + e
//                                       vector elem e>=8 -> K = 16 + half*8 + (e-8)
//   B (32x16 f16): lane = half*16 + n ; vector elem e -> same K order as A (e<8: half*8 + e, e>=8: 16 + half*8 + (e-8))
//   C/D (16x16 f32): VGPR r -> row m = 8*half + r, column n.
// ---------------------------------------------------------------------------
__global__ __launch_bounds__(256) void gemm_fused(
    const _Float16* __restrict__ Au,
    const _Float16* __restrict__ Ai,
    const _Float16* __restrict__ Wuh,
    const _Float16* __restrict__ Wih,
    const float* __restrict__ bu,
    const float* __restrict__ bi,
    const float* __restrict__ idisc,
    float* __restrict__ out) {
    const int wave = threadIdx.x >> 5;
    const int lane = threadIdx.x & 31;
    const int half = lane >> 4;
    const int nl   = lane & 15;          // row index m for A loads, col n for B/C
    const int row_base = (blockIdx.x * 8 + wave) * 16;

    union Frag { uint4 q[2]; v16h v; };

    // Preload A fragments for the whole K=128 (4 K-steps, both matrices).
    v16h au[4], ai[4];
#pragma unroll
    for (int kt = 0; kt < 4; ++kt) {
        const _Float16* ap = Au + (size_t)(row_base + nl) * KDIM + kt * 32 + half * 8;
        Frag fa;
        fa.q[0] = *(const uint4*)ap;          // K offsets +0..7
        fa.q[1] = *(const uint4*)(ap + 16);   // K offsets +16..23
        au[kt] = fa.v;
        const _Float16* aip = Ai + (size_t)(row_base + nl) * KDIM + kt * 32 + half * 8;
        Frag fi;
        fi.q[0] = *(const uint4*)aip;
        fi.q[1] = *(const uint4*)(aip + 16);
        ai[kt] = fi.v;
    }

    float acc[8];
#pragma unroll
    for (int r = 0; r < 8; ++r) acc[r] = 0.0f;

    for (int nt = 0; nt < HID / 16; ++nt) {
        const int n_base = nt * 16;
        v8f cu = {};
        v8f ci = {};
#pragma unroll
        for (int kt = 0; kt < 4; ++kt) {
            const _Float16* bup = Wuh + (size_t)(n_base + nl) * KDIM + kt * 32 + half * 8;
            Frag fb;
            fb.q[0] = *(const uint4*)bup;
            fb.q[1] = *(const uint4*)(bup + 16);
            cu = __builtin_amdgcn_wmma_f32_16x16x32_f16(
                false, au[kt], false, fb.v, (short)0, cu, false, false);
            const _Float16* bip = Wih + (size_t)(n_base + nl) * KDIM + kt * 32 + half * 8;
            Frag gb;
            gb.q[0] = *(const uint4*)bip;
            gb.q[1] = *(const uint4*)(bip + 16);
            ci = __builtin_amdgcn_wmma_f32_16x16x32_f16(
                false, ai[kt], false, gb.v, (short)0, ci, false, false);
        }
        const float bub = bu[n_base + nl];
        const float bib = bi[n_base + nl];
#pragma unroll
        for (int r = 0; r < 8; ++r) {
            float uf = tanhf_(cu[r] + bub);
            float vf = sigmoidf_(ci[r] + bib);
            acc[r] += uf * vf;
        }
    }

    // Reduce across the 16 lanes of each half-wave (columns of this row set).
#pragma unroll
    for (int mask = 1; mask <= 8; mask <<= 1) {
#pragma unroll
        for (int r = 0; r < 8; ++r)
            acc[r] += __shfl_xor(acc[r], mask, 32);
    }

    if (nl == 0) {
#pragma unroll
        for (int r = 0; r < 8; ++r) {
            int row = row_base + half * 8 + r;
            out[row] = sigmoidf_(acc[r] - idisc[row]);
        }
    }
}

// ---------------------------------------------------------------------------
// Workspace layout (bytes):
//   Au    : [0,        4 MiB)   16384*128 f16
//   Ai    : [4 MiB,    8 MiB)   16384*128 f16
//   Wuh   : [8 MiB,    +64 KiB) 256*128 f16
//   Wih   : [+64 KiB,  +64 KiB)
//   idisc : [+128 KiB, +64 KiB) 16384 f32
// Total ~8.6 MiB.
// ---------------------------------------------------------------------------
extern "C" void kernel_launch(void* const* d_in, const int* in_sizes, int n_in,
                              void* d_out, int out_size, void* d_ws, size_t ws_size,
                              hipStream_t stream) {
    const int*   user_ids    = (const int*)d_in[0];
    const int*   item_ids    = (const int*)d_in[1];
    const float* item_know   = (const float*)d_in[2];
    const float* priori      = (const float*)d_in[3];
    const float* condi_p     = (const float*)d_in[4];
    const float* condi_n     = (const float*)d_in[5];
    const float* item_diff_w = (const float*)d_in[6];
    const float* item_disc_w = (const float*)d_in[7];
    const float* Wu          = (const float*)d_in[8];
    const float* bu          = (const float*)d_in[9];
    const float* Wi          = (const float*)d_in[10];
    const float* bi          = (const float*)d_in[11];
    float* out = (float*)d_out;

    char* ws = (char*)d_ws;
    _Float16* Au    = (_Float16*)(ws);
    _Float16* Ai    = (_Float16*)(ws + (size_t)4 * 1024 * 1024);
    _Float16* Wuh   = (_Float16*)(ws + (size_t)8 * 1024 * 1024);
    _Float16* Wih   = (_Float16*)(ws + (size_t)8 * 1024 * 1024 + 64 * 1024);
    float*    idisc = (float*)   (ws + (size_t)8 * 1024 * 1024 + 128 * 1024);

    cvt_kernel<<<(HID * KDIM + 255) / 256, 256, 0, stream>>>(Wu, Wi, Wuh, Wih);
    prep_kernel<<<BATCH_N / 256, 256, 0, stream>>>(
        user_ids, item_ids, item_know, priori, condi_p, condi_n,
        item_diff_w, item_disc_w, Au, Ai, idisc);
    gemm_fused<<<BATCH_N / 16 / 8, 256, 0, stream>>>(
        Au, Ai, Wuh, Wih, bu, bi, idisc, out);
}
